// _GTSEdgeTransformerLayer_62002147885261
// MI455X (gfx1250) — compile-verified, matrix operand order checked by reading, not run on hardware
//
#include <hip/hip_runtime.h>
#include <hip/hip_bf16.h>

typedef _Float16 half_t;
typedef __attribute__((ext_vector_type(16))) _Float16 v16h;
typedef __attribute__((ext_vector_type(8)))  _Float16 v8h;
typedef __attribute__((ext_vector_type(8)))  float    v8f;

#define BB   4
#define NN   64
#define EE   128
#define HH   8
#define DKK  16
#define ROWS (BB * NN * NN)   // 16384
#define PJ   (4 * EE)         // 512 columns in fused projection GEMM

// ---------------------------------------------------------------------------
// Weight pack: f32 (K,N) row-major -> f16 WMMA B-fragment layout.
// dst linear index i = ((kt*N + n)*2 + hb)*16 + e ; src K = kt*32 + (e<8?0:16) + hb*8 + (e&7)
// ---------------------------------------------------------------------------
__global__ void pack_b_kernel(const float* __restrict__ w, half_t* __restrict__ dst,
                              int K, int N) {
    int i = blockIdx.x * blockDim.x + threadIdx.x;
    if (i >= K * N) return;
    int e  = i & 15;
    int hb = (i >> 4) & 1;
    int n  = (i >> 5) % N;
    int kt = i / (N * 32);
    int k  = kt * 32 + ((e >> 3) << 4) + hb * 8 + (e & 7);
    dst[i] = (half_t)w[(size_t)k * N + n];
}

// Fused pack of [W_lk | W_rk | W_lv | W_rv] -> (128 x 512) fragment layout
__global__ void pack_cat_kernel(const float* __restrict__ w0, const float* __restrict__ w1,
                                const float* __restrict__ w2, const float* __restrict__ w3,
                                half_t* __restrict__ dst) {
    int i = blockIdx.x * blockDim.x + threadIdx.x;
    if (i >= EE * PJ) return;
    int e  = i & 15;
    int hb = (i >> 4) & 1;
    int n  = (i >> 5) & (PJ - 1);
    int kt = i >> 14;               // PJ*32 = 16384 elements per k-tile
    int k  = kt * 32 + ((e >> 3) << 4) + hb * 8 + (e & 7);
    const float* w = (n < 128) ? w0 : (n < 256) ? w1 : (n < 384) ? w2 : w3;
    dst[i] = (half_t)w[(size_t)k * EE + (n & 127)];
}

// ---------------------------------------------------------------------------
// Row LayerNorm over E=128 (one wave per row), optional residual input,
// optional f32 and f16 outputs.
// ---------------------------------------------------------------------------
__global__ void ln_rows_kernel(const float* __restrict__ x, const float* __restrict__ resid,
                               const float* __restrict__ g, const float* __restrict__ bta,
                               float* __restrict__ of, half_t* __restrict__ oh, int rows) {
    int wave = threadIdx.x >> 5;
    int lane = threadIdx.x & 31;
    int row  = blockIdx.x * (blockDim.x >> 5) + wave;
    if (row >= rows) return;

    float4 v = ((const float4*)(x + (size_t)row * EE))[lane];
    if (resid) {
        float4 r = ((const float4*)(resid + (size_t)row * EE))[lane];
        v.x += r.x; v.y += r.y; v.z += r.z; v.w += r.w;
    }
    float s  = v.x + v.y + v.z + v.w;
    float ss = v.x * v.x + v.y * v.y + v.z * v.z + v.w * v.w;
    #pragma unroll
    for (int off = 16; off > 0; off >>= 1) {
        s  += __shfl_xor(s,  off, 32);
        ss += __shfl_xor(ss, off, 32);
    }
    float mean = s * (1.0f / EE);
    float var  = ss * (1.0f / EE) - mean * mean;
    float inv  = rsqrtf(var + 1e-5f);

    float4 gg = ((const float4*)g)[lane];
    float4 bb = ((const float4*)bta)[lane];
    float o0 = (v.x - mean) * inv * gg.x + bb.x;
    float o1 = (v.y - mean) * inv * gg.y + bb.y;
    float o2 = (v.z - mean) * inv * gg.z + bb.z;
    float o3 = (v.w - mean) * inv * gg.w + bb.w;

    if (of) {
        float4 o; o.x = o0; o.y = o1; o.z = o2; o.w = o3;
        ((float4*)(of + (size_t)row * EE))[lane] = o;
    }
    if (oh) {
        half_t* p = oh + (size_t)row * EE + lane * 4;
        p[0] = (half_t)o0; p[1] = (half_t)o1; p[2] = (half_t)o2; p[3] = (half_t)o3;
    }
}

// ---------------------------------------------------------------------------
// WMMA GEMM, register-blocked: each wave computes a 16(M) x 64(N) strip with
// 4 accumulators; one A fragment feeds 4 v_wmma per K-step (K stepped by 32).
// ---------------------------------------------------------------------------
__device__ __forceinline__ void store_tile(v8f c, int row0, int col, int hb, int N,
                                           const float* __restrict__ bias, int relu,
                                           float* __restrict__ Cf, half_t* __restrict__ Ch) {
    float bv = bias ? bias[col] : 0.0f;
    #pragma unroll
    for (int r = 0; r < 8; ++r) {
        int row = row0 + r + hb * 8;
        float v = c[r] + bv;
        if (relu) v = fmaxf(v, 0.0f);
        size_t idx = (size_t)row * N + col;
        if (Cf) Cf[idx] = v;
        if (Ch) Ch[idx] = (half_t)v;
    }
}

__global__ void gemm_wmma_kernel(const half_t* __restrict__ A, const half_t* __restrict__ Bp,
                                 float* __restrict__ Cf, half_t* __restrict__ Ch,
                                 const float* __restrict__ bias,
                                 int M, int N, int K, int relu) {
    const int lane = threadIdx.x & 31;
    const int wave = threadIdx.x >> 5;
    const int stripsN = N >> 6;                       // 64-wide N strips
    const int strip = blockIdx.x * (blockDim.x >> 5) + wave;
    if (strip >= (M >> 4) * stripsN) return;          // wave-uniform branch

    const int tr = strip / stripsN, tc = strip % stripsN;
    const int row0 = tr << 4, col0 = tc << 6;
    const int m  = lane & 15;
    const int hb = lane >> 4;

    v8f c0 = {}, c1 = {}, c2 = {}, c3 = {};
    const half_t* arow = A + (size_t)(row0 + m) * K + hb * 8;
    const int ksteps = K >> 5;

    for (int kt = 0; kt < ksteps; ++kt) {
        __builtin_prefetch(arow + 64, 0, 1);          // global_prefetch_b8 next A line
        v8h a0 = *(const v8h*)(arow);                 // K = 32*kt + hb*8 + [0..7]
        v8h a1 = *(const v8h*)(arow + 16);            // K = 32*kt + 16 + hb*8 + [0..7]
        v16h a = __builtin_shufflevector(a0, a1, 0, 1, 2, 3, 4, 5, 6, 7,
                                                 8, 9, 10, 11, 12, 13, 14, 15);
        const half_t* bp = Bp + (((size_t)kt * N + col0 + m) * 2 + hb) * 16;
        v16h b0 = *(const v16h*)(bp);                 // n = col0 + m
        v16h b1 = *(const v16h*)(bp + 512);           // n = col0 + 16 + m
        v16h b2 = *(const v16h*)(bp + 1024);          // n = col0 + 32 + m
        v16h b3 = *(const v16h*)(bp + 1536);          // n = col0 + 48 + m
        c0 = __builtin_amdgcn_wmma_f32_16x16x32_f16(false, a, false, b0, (short)0, c0, false, false);
        c1 = __builtin_amdgcn_wmma_f32_16x16x32_f16(false, a, false, b1, (short)0, c1, false, false);
        c2 = __builtin_amdgcn_wmma_f32_16x16x32_f16(false, a, false, b2, (short)0, c2, false, false);
        c3 = __builtin_amdgcn_wmma_f32_16x16x32_f16(false, a, false, b3, (short)0, c3, false, false);
        arow += 32;
    }

    store_tile(c0, row0, col0 +  0 + m, hb, N, bias, relu, Cf, Ch);
    store_tile(c1, row0, col0 + 16 + m, hb, N, bias, relu, Cf, Ch);
    store_tile(c2, row0, col0 + 32 + m, hb, N, bias, relu, Cf, Ch);
    store_tile(c3, row0, col0 + 48 + m, hb, N, bias, relu, Cf, Ch);
}

// ---------------------------------------------------------------------------
// Fused triangle attention for one (b, h, x): scores -> mask -> softmax(a) ->
// 3-way ctx einsum. Scores live only in LDS (16 KB), never hit HBM.
// proj row-major ROWS x 512 : [lk | rk | lv | rv], head slice = h*16.
// ---------------------------------------------------------------------------
__global__ void tri_attn_kernel(const float* __restrict__ proj,
                                const unsigned char* __restrict__ mask,
                                half_t* __restrict__ ctx_h) {
    __shared__ float s_lk[NN][DKK];
    __shared__ float s_lv[NN][DKK];
    __shared__ float S[NN][NN + 1];

    const int bid = blockIdx.x;
    const int x = bid & (NN - 1);
    const int h = (bid >> 6) & (HH - 1);
    const int b = bid >> 9;
    const int t = threadIdx.x;              // 256 threads

    const size_t rowbase_x = ((size_t)b * NN + x) * NN;   // rows (b,x,a)
    const int hoff = h * DKK;

    // stage lk / lv slabs for this (b, x, h)
    for (int i = t; i < NN * DKK; i += 256) {
        int a = i >> 4, d = i & 15;
        const float* pr = proj + (rowbase_x + a) * PJ;
        s_lk[a][d] = pr[hoff + d];
        s_lv[a][d] = pr[2 * EE + hoff + d];
    }
    __syncthreads();

    // scores S[a][y] = lk[x,a,:] . rk[a,y,:] / sqrt(DK), masked
    const size_t mbase = (((size_t)b * NN + x) * NN) * NN;
    for (int i = t; i < NN * NN; i += 256) {
        int a = i >> 6, y = i & 63;
        const float* rk = proj + (((size_t)b * NN + a) * NN + y) * PJ + EE + hoff;
        float acc = 0.0f;
        #pragma unroll
        for (int d = 0; d < DKK; ++d) acc += s_lk[a][d] * rk[d];
        acc *= 0.25f;                       // 1/sqrt(16)
        if (!mask[mbase + (size_t)a * NN + y]) acc = -1e9f;
        S[a][y] = acc;
    }
    __syncthreads();

    // softmax over a, one column y per thread
    if (t < NN) {
        int y = t;
        float mx = -3.402823466e+38f;
        for (int a = 0; a < NN; ++a) mx = fmaxf(mx, S[a][y]);
        float sum = 0.0f;
        for (int a = 0; a < NN; ++a) { float e = __expf(S[a][y] - mx); S[a][y] = e; sum += e; }
        float r = 1.0f / sum;
        for (int a = 0; a < NN; ++a) S[a][y] *= r;
    }
    __syncthreads();

    // ctx[y][d] = sum_a att[a][y] * lv[a][d] * rv[a,y,d]
    const int y  = t & 63;
    const int dq = t >> 6;                  // 4 d-values per thread
    float acc0 = 0, acc1 = 0, acc2 = 0, acc3 = 0;
    for (int a = 0; a < NN; ++a) {
        float att = S[a][y];
        const float4 rv = *(const float4*)(proj + (((size_t)b * NN + a) * NN + y) * PJ
                                           + 3 * EE + hoff + dq * 4);
        acc0 += att * s_lv[a][dq * 4 + 0] * rv.x;
        acc1 += att * s_lv[a][dq * 4 + 1] * rv.y;
        acc2 += att * s_lv[a][dq * 4 + 2] * rv.z;
        acc3 += att * s_lv[a][dq * 4 + 3] * rv.w;
    }
    half_t* o = ctx_h + (rowbase_x + y) * EE + hoff + dq * 4;
    o[0] = (half_t)acc0; o[1] = (half_t)acc1; o[2] = (half_t)acc2; o[3] = (half_t)acc3;
}

// ---------------------------------------------------------------------------
extern "C" void kernel_launch(void* const* d_in, const int* in_sizes, int n_in,
                              void* d_out, int out_size, void* d_ws, size_t ws_size,
                              hipStream_t stream) {
    const float* x_in = (const float*)d_in[0];
    const unsigned char* mask = (const unsigned char*)d_in[1];
    const float* W_lk = (const float*)d_in[2];
    const float* W_rk = (const float*)d_in[3];
    const float* W_lv = (const float*)d_in[4];
    const float* W_rv = (const float*)d_in[5];
    const float* W_o  = (const float*)d_in[6];
    const float* ln_g = (const float*)d_in[7];
    const float* ln_b = (const float*)d_in[8];
    const float* W1   = (const float*)d_in[9];
    const float* b1   = (const float*)d_in[10];
    const float* W2   = (const float*)d_in[11];
    const float* b2   = (const float*)d_in[12];
    const float* na_g = (const float*)d_in[13];
    const float* na_b = (const float*)d_in[14];
    const float* n_g  = (const float*)d_in[15];
    const float* n_b  = (const float*)d_in[16];
    float* out = (float*)d_out;

    // workspace carving (all offsets 256B aligned)
    char* ws = (char*)d_ws;
    size_t off = 0;
    auto take = [&](size_t bytes) { void* p = ws + off; off += (bytes + 255) & ~(size_t)255; return p; };
    half_t* x_h      = (half_t*)take((size_t)ROWS * EE * 2);       // 4 MB
    float*  proj     = (float*) take((size_t)ROWS * PJ * 4);       // 32 MB
    half_t* ctx_h    = (half_t*)take((size_t)ROWS * EE * 2);       // 4 MB
    float*  attn_out = (float*) take((size_t)ROWS * EE * 4);       // 8 MB
    float*  y_f      = (float*) take((size_t)ROWS * EE * 4);       // 8 MB
    half_t* y_h      = (half_t*)take((size_t)ROWS * EE * 2);       // 4 MB
    half_t* h1_h     = (half_t*)take((size_t)ROWS * 2 * EE * 2);   // 8 MB
    float*  h2_f     = (float*) take((size_t)ROWS * EE * 4);       // 8 MB
    half_t* Wcat_p   = (half_t*)take((size_t)EE * PJ * 2);
    half_t* Wo_p     = (half_t*)take((size_t)EE * EE * 2);
    half_t* W1_p     = (half_t*)take((size_t)EE * 2 * EE * 2);
    half_t* W2_p     = (half_t*)take((size_t)2 * EE * EE * 2);
    (void)ws_size; (void)in_sizes; (void)n_in; (void)out_size;

    // 1) weight packing into WMMA fragment layout
    pack_cat_kernel<<<(EE * PJ + 255) / 256, 256, 0, stream>>>(W_lk, W_rk, W_lv, W_rv, Wcat_p);
    pack_b_kernel<<<(EE * EE + 255) / 256, 256, 0, stream>>>(W_o, Wo_p, EE, EE);
    pack_b_kernel<<<(EE * 2 * EE + 255) / 256, 256, 0, stream>>>(W1, W1_p, EE, 2 * EE);
    pack_b_kernel<<<(2 * EE * EE + 255) / 256, 256, 0, stream>>>(W2, W2_p, 2 * EE, EE);

    // 2) pre-norm -> f16
    ln_rows_kernel<<<ROWS / 8, 256, 0, stream>>>(x_in, nullptr, ln_g, ln_b,
                                                 nullptr, x_h, ROWS);

    // 3) fused projections: (16384 x 128) @ (128 x 512) -> proj f32
    {
        int strips = (ROWS / 16) * (PJ / 64);          // 8192
        gemm_wmma_kernel<<<strips / 8, 256, 0, stream>>>(x_h, Wcat_p, proj, nullptr,
                                                         nullptr, ROWS, PJ, EE, 0);
    }

    // 4) fused triangle attention -> ctx f16
    tri_attn_kernel<<<BB * HH * NN, 256, 0, stream>>>(proj, mask, ctx_h);

    // 5) output projection
    {
        int strips = (ROWS / 16) * (EE / 64);          // 2048
        gemm_wmma_kernel<<<strips / 8, 256, 0, stream>>>(ctx_h, Wo_p, attn_out, nullptr,
                                                         nullptr, ROWS, EE, EE, 0);
    }

    // 6) y = LN(x_in + attn_out)
    ln_rows_kernel<<<ROWS / 8, 256, 0, stream>>>(attn_out, x_in, na_g, na_b,
                                                 y_f, y_h, ROWS);

    // 7) FFN1 + bias + relu -> f16
    {
        int strips = (ROWS / 16) * (2 * EE / 64);      // 4096
        gemm_wmma_kernel<<<strips / 8, 256, 0, stream>>>(y_h, W1_p, nullptr, h1_h,
                                                         b1, ROWS, 2 * EE, EE, 1);
    }

    // 8) FFN2 + bias -> f32
    {
        int strips = (ROWS / 16) * (EE / 64);          // 2048
        gemm_wmma_kernel<<<strips / 8, 256, 0, stream>>>(h1_h, W2_p, h2_f, nullptr,
                                                         b2, ROWS, EE, 2 * EE, 0);
    }

    // 9) out = LN(h2 + y)
    ln_rows_kernel<<<ROWS / 8, 256, 0, stream>>>(h2_f, y_f, n_g, n_b,
                                                 out, nullptr, ROWS);
}
